// mamba_Branches_74294344286681
// MI455X (gfx1250) — compile-verified, matrix operand order checked by reading, not run on hardware
//
#include <hip/hip_runtime.h>
#include <hip/hip_bf16.h>
#include <cstddef>
#include <cstdint>

// ---------------------------------------------------------------------------
// Types for CDNA5 WMMA (wave32, 16x16x32 bf16 -> f32)
// ---------------------------------------------------------------------------
typedef __bf16 bf16_t;
typedef bf16_t v16bf __attribute__((ext_vector_type(16)));
typedef bf16_t v8bf  __attribute__((ext_vector_type(8)));
typedef float  v8f   __attribute__((ext_vector_type(8)));

#if __has_builtin(__builtin_amdgcn_global_load_async_to_lds_b32) && \
    __has_builtin(__builtin_amdgcn_s_wait_asynccnt)
#define USE_ASYNC_LDS 1
typedef __attribute__((address_space(1))) int as1_int;
typedef __attribute__((address_space(3))) int as3_int;
#else
#define USE_ASYNC_LDS 0
#endif

static __device__ __forceinline__ float dev_sigmoid(float x) {
  return 1.0f / (1.0f + __expf(-x));
}

// ---------------------------------------------------------------------------
// Conversion kernels
// ---------------------------------------------------------------------------
__global__ void k_f32_to_bf16_rows(const float* __restrict__ src, int srcStride,
                                   bf16_t* __restrict__ dst, int cols, long long total) {
  for (long long i = blockIdx.x * (long long)blockDim.x + threadIdx.x; i < total;
       i += (long long)gridDim.x * blockDim.x) {
    long long r = i / cols;
    int c = (int)(i - r * cols);
    dst[i] = (bf16_t)src[r * (long long)srcStride + c];
  }
}

// w: [K,N] row-major f32  ->  bt: [N,K] row-major bf16
__global__ void k_transpose_to_bf16(const float* __restrict__ w, bf16_t* __restrict__ bt,
                                    int K, int N, long long total) {
  for (long long i = blockIdx.x * (long long)blockDim.x + threadIdx.x; i < total;
       i += (long long)gridDim.x * blockDim.x) {
    long long k = i / N;
    int n = (int)(i - k * N);
    bt[(long long)n * K + k] = (bf16_t)w[i];
  }
}

// ---------------------------------------------------------------------------
// WMMA GEMM: C[M,N] = act(A[M,K] * Bt[N,K]^T + bias) (+= existing C if accum)
// One wave per 16x16 tile. K multiple of 32, M,N multiples of 16.
// Per-lane fragment = two contiguous 16B chunks -> explicit b128 loads.
// act: 0 = none, 1 = softplus
// ---------------------------------------------------------------------------
__global__ __launch_bounds__(32)
void k_gemm_wmma_bf16(const bf16_t* __restrict__ A, const bf16_t* __restrict__ Bt,
                      const float* __restrict__ bias, float* __restrict__ C,
                      int M, int N, int K, int act, int accum) {
  const int lane = threadIdx.x;          // 0..31
  const int n0 = blockIdx.x * 16;
  const int m0 = blockIdx.y * 16;
  const int r = lane & 15;
  const int g = lane >> 4;               // lane group 0/1

  const bf16_t* __restrict__ Arow = A + (size_t)(m0 + r) * K;
  const bf16_t* __restrict__ Brow = Bt + (size_t)(n0 + r) * K;

  v8f acc = {0.f, 0.f, 0.f, 0.f, 0.f, 0.f, 0.f, 0.f};

  for (int k0 = 0; k0 < K; k0 += 32) {
    if (k0 + 32 < K) {
      __builtin_prefetch(Arow + k0 + 32, 0, 0);   // global_prefetch_b8
      __builtin_prefetch(Brow + k0 + 32, 0, 0);
    }
    // CDNA5 ISA 7.12.2: lane-group g holds K = g*8+{0..7} in VGPR0-3 and
    // K = 16+g*8+{0..7} in VGPR4-7; both are 16B-aligned contiguous chunks.
    const v8bf* __restrict__ Ac = (const v8bf*)(Arow + k0);
    const v8bf* __restrict__ Bc = (const v8bf*)(Brow + k0);
    const v8bf a0 = Ac[g];
    const v8bf a1 = Ac[2 + g];
    const v8bf b0 = Bc[g];
    const v8bf b1 = Bc[2 + g];
    const v16bf a = __builtin_shufflevector(a0, a1, 0, 1, 2, 3, 4, 5, 6, 7,
                                            8, 9, 10, 11, 12, 13, 14, 15);
    const v16bf b = __builtin_shufflevector(b0, b1, 0, 1, 2, 3, 4, 5, 6, 7,
                                            8, 9, 10, 11, 12, 13, 14, 15);
    acc = __builtin_amdgcn_wmma_f32_16x16x32_bf16(
        /*neg_a=*/false, a, /*neg_b=*/false, b,
        /*c_mod=*/(short)0, acc, /*reuse_a=*/false, /*reuse_b=*/false);
  }

  const int n = n0 + r;
  const float bv = bias ? bias[n] : 0.f;
#pragma unroll
  for (int q = 0; q < 8; ++q) {
    const int m = m0 + g * 8 + q;              // C/D layout: lanes16-31 hold M+8
    float val = acc[q] + bv;
    if (act == 1) val = (val > 20.f) ? val : log1pf(__expf(val));  // softplus
    const size_t off = (size_t)m * N + n;
    if (accum) val += C[off];
    C[off] = val;
  }
}

// ---------------------------------------------------------------------------
// LayerNorm over last dim; one block per token, blockDim == dim (64 or 256)
// ---------------------------------------------------------------------------
__global__ void k_layernorm(const float* __restrict__ x, const float* __restrict__ g,
                            const float* __restrict__ b, float* __restrict__ out, int dim) {
  __shared__ float red[256];
  const int t = blockIdx.x;
  const int i = threadIdx.x;
  const float v = x[(size_t)t * dim + i];
  red[i] = v;
  __syncthreads();
  for (int s = dim >> 1; s > 0; s >>= 1) {
    if (i < s) red[i] += red[i + s];
    __syncthreads();
  }
  const float mu = red[0] * (1.0f / dim);
  __syncthreads();
  const float dv = v - mu;
  red[i] = dv * dv;
  __syncthreads();
  for (int s = dim >> 1; s > 0; s >>= 1) {
    if (i < s) red[i] += red[i + s];
    __syncthreads();
  }
  const float var = red[0] * (1.0f / dim);
  out[(size_t)t * dim + i] = dv * rsqrtf(var + 1e-5f) * g[i] + b[i];
}

// ---------------------------------------------------------------------------
// Causal depthwise conv (K=4) + SiLU.  xi lives in first di cols of xz[tok,2*di].
// ---------------------------------------------------------------------------
__global__ void k_dwconv_silu(const float* __restrict__ xz, const float* __restrict__ wc,
                              const float* __restrict__ bc, float* __restrict__ xcv,
                              int L, int di, long long total) {
  const int rs = 2 * di;
  for (long long i = blockIdx.x * (long long)blockDim.x + threadIdx.x; i < total;
       i += (long long)gridDim.x * blockDim.x) {
    const int d = (int)(i % di);
    const long long tok = i / di;
    const int l = (int)(tok % L);
    const long long bL = tok - l;
    float acc = bc[d];
#pragma unroll
    for (int k = 0; k < 4; ++k) {
      const int ll = l - 3 + k;
      if (ll >= 0) acc += wc[d * 4 + k] * xz[(bL + ll) * rs + d];
    }
    xcv[i] = acc * dev_sigmoid(acc);
  }
}

// y *= silu(z), z in cols [di,2di) of xz
__global__ void k_gate_silu(float* __restrict__ y, const float* __restrict__ xz,
                            int di, long long total) {
  for (long long i = blockIdx.x * (long long)blockDim.x + threadIdx.x; i < total;
       i += (long long)gridDim.x * blockDim.x) {
    const int d = (int)(i % di);
    const long long tok = i / di;
    const float zv = xz[tok * (2LL * di) + di + d];
    y[i] *= zv * dev_sigmoid(zv);
  }
}

// ---------------------------------------------------------------------------
// Selective scan. Block = 128 threads = one 128-channel chunk of one batch.
// Per-thread state h[NS] in registers; B/C rows staged into LDS via the
// gfx1250 async global->LDS path (ASYNCcnt / s_wait_asynccnt) when available.
// ---------------------------------------------------------------------------
template <int NS>
__global__ __launch_bounds__(128)
void k_scan(const float* __restrict__ xc, const float* __restrict__ dt,
            const float* __restrict__ dbc, int dbcN,
            const float* __restrict__ A_log, const float* __restrict__ Dp,
            float* __restrict__ y, int L, int di) {
  const int chunks = di >> 7;
  const int b = blockIdx.x / chunks;
  const int ch = blockIdx.x - b * chunks;
  const int d = (ch << 7) + threadIdx.x;
  __shared__ float BC[2 * NS];           // [0,NS): B row, [NS,2NS): C row
  float Arow[NS], h[NS];
#pragma unroll
  for (int n = 0; n < NS; ++n) {
    Arow[n] = -__expf(A_log[(size_t)d * NS + n]);
    h[n] = 0.f;
  }
  const float Dd = Dp[d];
  for (int l = 0; l < L; ++l) {
    const size_t t = (size_t)b * L + l;
#if USE_ASYNC_LDS
    if ((int)threadIdx.x < 2 * NS) {
      const float* src = &dbc[t * dbcN + 32 + threadIdx.x];
      __builtin_amdgcn_global_load_async_to_lds_b32(
          (as1_int*)src, (as3_int*)&BC[threadIdx.x],
          /*imm offset*/ 0, /*cpol*/ 0);
    }
    __builtin_amdgcn_s_wait_asynccnt(0);
#else
    if ((int)threadIdx.x < 2 * NS)
      BC[threadIdx.x] = dbc[t * dbcN + 32 + threadIdx.x];
#endif
    if (l + 1 < L) __builtin_prefetch(&dbc[(t + 1) * dbcN + 32], 0, 0);
    __syncthreads();
    const float xv = xc[t * di + d];
    const float dtv = dt[t * di + d];
    const float dx = dtv * xv;
    float acc = 0.f;
#pragma unroll
    for (int n = 0; n < NS; ++n) {
      h[n] = __expf(dtv * Arow[n]) * h[n] + dx * BC[n];
      acc += h[n] * BC[NS + n];
    }
    y[t * di + d] = acc + xv * Dd;
    __syncthreads();
  }
}

// ---------------------------------------------------------------------------
// Convolutions / reshapes
// ---------------------------------------------------------------------------
__global__ void k_conv3x3_reflect(const float* __restrict__ in, const float* __restrict__ w,
                                  const float* __restrict__ bias, float* __restrict__ out,
                                  int B, int CI, int CO, int H, int W, long long total) {
  for (long long i = blockIdx.x * (long long)blockDim.x + threadIdx.x; i < total;
       i += (long long)gridDim.x * blockDim.x) {
    int ow = (int)(i % W);
    long long t = i / W;
    int oh = (int)(t % H); t /= H;
    int co = (int)(t % CO);
    int b = (int)(t / CO);
    float acc = bias[co];
    for (int ci = 0; ci < CI; ++ci)
      for (int ky = 0; ky < 3; ++ky) {
        int ih = oh - 1 + ky;
        ih = (ih < 0) ? -ih : ((ih >= H) ? 2 * H - 2 - ih : ih);
        for (int kx = 0; kx < 3; ++kx) {
          int iw = ow - 1 + kx;
          iw = (iw < 0) ? -iw : ((iw >= W) ? 2 * W - 2 - iw : iw);
          acc += w[((co * CI + ci) * 3 + ky) * 3 + kx] *
                 in[((size_t)(b * CI + ci) * H + ih) * W + iw];
        }
      }
    out[i] = acc;
  }
}

// down1: [8,16,256,256] -k4s4-> token-major seq [8,4096,64]
__global__ void k_down1(const float* __restrict__ in, const float* __restrict__ w,
                        const float* __restrict__ bias, float* __restrict__ seq,
                        long long total) {
  for (long long i = blockIdx.x * (long long)blockDim.x + threadIdx.x; i < total;
       i += (long long)gridDim.x * blockDim.x) {
    const int co = (int)(i & 63);
    const int tok = (int)((i >> 6) & 4095);
    const int b = (int)(i >> 18);
    const int oh = tok >> 6, ow = tok & 63;
    float acc = bias[co];
    for (int ci = 0; ci < 16; ++ci)
      for (int ky = 0; ky < 4; ++ky)
        for (int kx = 0; kx < 4; ++kx)
          acc += w[((co * 16 + ci) * 4 + ky) * 4 + kx] *
                 in[((size_t)(b * 16 + ci) * 256 + (oh * 4 + ky)) * 256 + (ow * 4 + kx)];
    seq[i] = acc;
  }
}

// down2 im2col: seq64 [8,4096,64] -> bf16 A [8192,256]; k = (dy*2+dx)*64+ci
__global__ void k_im2col_down2(const float* __restrict__ seq64, bf16_t* __restrict__ a,
                               long long total) {
  for (long long i = blockIdx.x * (long long)blockDim.x + threadIdx.x; i < total;
       i += (long long)gridDim.x * blockDim.x) {
    const int k = (int)(i & 255);
    const int row = (int)(i >> 8);
    const int b = row >> 10, tok = row & 1023;
    const int oh = tok >> 5, ow = tok & 31;
    const int dy = k >> 7, dx = (k >> 6) & 1, ci = k & 63;
    const int ih = 2 * oh + dy, iw = 2 * ow + dx;
    a[i] = (bf16_t)seq64[((size_t)(b * 4096 + ih * 64 + iw)) * 64 + ci];
  }
}

// down2 weight [256,64,2,2] -> Bt [256,256] matching im2col k-order
__global__ void k_pack_down2_w(const float* __restrict__ w, bf16_t* __restrict__ bt,
                               long long total) {
  for (long long i = blockIdx.x * (long long)blockDim.x + threadIdx.x; i < total;
       i += (long long)gridDim.x * blockDim.x) {
    const int co = (int)(i >> 8);
    const int k = (int)(i & 255);
    const int dy = k >> 7, dx = (k >> 6) & 1, ci = k & 63;
    bt[i] = (bf16_t)w[((co * 64 + ci) * 2 + dy) * 2 + dx];
  }
}

__global__ void k_seq_to_nchw(const float* __restrict__ seq, float* __restrict__ out,
                              int C, int H, int W, long long total) {
  for (long long i = blockIdx.x * (long long)blockDim.x + threadIdx.x; i < total;
       i += (long long)gridDim.x * blockDim.x) {
    int w = (int)(i % W);
    long long t = i / W;
    int h = (int)(t % H); t /= H;
    int c = (int)(t % C);
    int b = (int)(t / C);
    out[i] = seq[((size_t)b * H * W + h * W + w) * C + c];
  }
}

// pixel_shuffle r=2 in token-major: [8,1024,256] -> [8,4096,64]
__global__ void k_ps2_seq(const float* __restrict__ in, float* __restrict__ out,
                          long long total) {
  for (long long i = blockIdx.x * (long long)blockDim.x + threadIdx.x; i < total;
       i += (long long)gridDim.x * blockDim.x) {
    const int c2 = (int)(i & 63);
    const int t2 = (int)((i >> 6) & 4095);
    const int b = (int)(i >> 18);
    const int oh = t2 >> 6, ow = t2 & 63;
    const int h = oh >> 1, r1 = oh & 1, w = ow >> 1, r2 = ow & 1;
    const int cin = c2 * 4 + r1 * 2 + r2;
    out[i] = in[((size_t)(b * 1024 + h * 32 + w)) * 256 + cin];
  }
}

// pixel_shuffle r=4, seq [8,4096,256] -> NCHW [8,16,256,256]
__global__ void k_ps4_nchw(const float* __restrict__ in, float* __restrict__ out,
                           long long total) {
  for (long long i = blockIdx.x * (long long)blockDim.x + threadIdx.x; i < total;
       i += (long long)gridDim.x * blockDim.x) {
    const int ow = (int)(i & 255);
    const int oh = (int)((i >> 8) & 255);
    const int c2 = (int)((i >> 16) & 15);
    const int b = (int)(i >> 20);
    const int h = oh >> 2, r1 = oh & 3, w = ow >> 2, r2 = ow & 3;
    const int cin = c2 * 16 + r1 * 4 + r2;
    out[i] = in[((size_t)(b * 4096 + h * 64 + w)) * 256 + cin];
  }
}

// ---------------------------------------------------------------------------
// Host-side helpers
// ---------------------------------------------------------------------------
static inline dim3 g1(long long n) {
  long long b = (n + 255) / 256;
  if (b > 1048576) b = 1048576;
  return dim3((unsigned)b);
}

static inline void launch_gemm(hipStream_t st, const bf16_t* A, const bf16_t* Bt,
                               const float* bias, float* C, int M, int N, int K,
                               int act, int accum) {
  dim3 grid(N / 16, M / 16);
  k_gemm_wmma_bf16<<<grid, 32, 0, st>>>(A, Bt, bias, C, M, N, K, act, accum);
}

// canonical mamba leaf order: A_log, D, b_conv, b_dt, b_in, b_out, ln_b, ln_g,
//                             w_conv, w_dt, w_in, w_out, w_xproj
struct MambaP { const float* p[13]; };

static void run_mamba(hipStream_t st, const MambaP& mp, float* s, int B, int L,
                      int dim, int ns, float* lnb, float* xz, float* xcv,
                      float* dbc, float* dtb, float* yb, bf16_t* A16, bf16_t* W16) {
  const float* A_log = mp.p[0];
  const float* Dp    = mp.p[1];
  const float* b_cv  = mp.p[2];
  const float* b_dt  = mp.p[3];
  const float* b_in  = mp.p[4];
  const float* b_out = mp.p[5];
  const float* ln_b  = mp.p[6];
  const float* ln_g  = mp.p[7];
  const float* w_cv  = mp.p[8];
  const float* w_dt  = mp.p[9];
  const float* w_in  = mp.p[10];
  const float* w_out = mp.p[11];
  const float* w_xp  = mp.p[12];

  const int di = 2 * dim;
  const int tokens = B * L;
  const int xzN = 2 * di;
  const int dbcN = 32 + 2 * ns;

  k_layernorm<<<tokens, dim, 0, st>>>(s, ln_g, ln_b, lnb, dim);

  long long tA = (long long)tokens * dim;
  k_f32_to_bf16_rows<<<g1(tA), 256, 0, st>>>(lnb, dim, A16, dim, tA);
  long long tW = (long long)dim * xzN;
  k_transpose_to_bf16<<<g1(tW), 256, 0, st>>>(w_in, W16, dim, xzN, tW);
  launch_gemm(st, A16, W16, b_in, xz, tokens, xzN, dim, 0, 0);

  long long tDI = (long long)tokens * di;
  k_dwconv_silu<<<g1(tDI), 256, 0, st>>>(xz, w_cv, b_cv, xcv, L, di, tDI);

  k_f32_to_bf16_rows<<<g1(tDI), 256, 0, st>>>(xcv, di, A16, di, tDI);
  tW = (long long)di * dbcN;
  k_transpose_to_bf16<<<g1(tW), 256, 0, st>>>(w_xp, W16, di, dbcN, tW);
  launch_gemm(st, A16, W16, nullptr, dbc, tokens, dbcN, di, 0, 0);

  long long tR = (long long)tokens * 32;
  k_f32_to_bf16_rows<<<g1(tR), 256, 0, st>>>(dbc, dbcN, A16, 32, tR);
  tW = (long long)32 * di;
  k_transpose_to_bf16<<<g1(tW), 256, 0, st>>>(w_dt, W16, 32, di, tW);
  launch_gemm(st, A16, W16, b_dt, dtb, tokens, di, 32, /*softplus*/ 1, 0);

  dim3 sg(B * (di / 128));
  if (ns == 16)
    k_scan<16><<<sg, 128, 0, st>>>(xcv, dtb, dbc, dbcN, A_log, Dp, yb, L, di);
  else
    k_scan<64><<<sg, 128, 0, st>>>(xcv, dtb, dbc, dbcN, A_log, Dp, yb, L, di);

  k_gate_silu<<<g1(tDI), 256, 0, st>>>(yb, xz, di, tDI);

  k_f32_to_bf16_rows<<<g1(tDI), 256, 0, st>>>(yb, di, A16, di, tDI);
  tW = (long long)di * dim;
  k_transpose_to_bf16<<<g1(tW), 256, 0, st>>>(w_out, W16, di, dim, tW);
  launch_gemm(st, A16, W16, b_out, s, tokens, dim, di, 0, /*accum residual*/ 1);
}

// ---------------------------------------------------------------------------
extern "C" void kernel_launch(void* const* d_in, const int* in_sizes, int n_in,
                              void* d_out, int out_size, void* d_ws, size_t ws_size,
                              hipStream_t stream) {
  (void)n_in; (void)out_size; (void)ws_size;
  const float* c_in = (const float*)d_in[0];

  // --- resolve flattening convention (sorted-key jax tree vs insertion order)
  const bool sorted = (in_sizes[1] == 16);  // conv_in.b first => sorted
  const float *ci_w, *ci_b, *co_w, *co_b, *d1_w, *d1_b, *d2_w, *d2_b;
  const float *u1_w, *u1_b, *u2_w, *u2_b;
  MambaP v256[2], v64d[2], v64u[2];
  if (sorted) {
    ci_b = (const float*)d_in[1];  ci_w = (const float*)d_in[2];
    co_b = (const float*)d_in[3];  co_w = (const float*)d_in[4];
    d1_b = (const float*)d_in[5];  d1_w = (const float*)d_in[6];
    d2_b = (const float*)d_in[7];  d2_w = (const float*)d_in[8];
    u1_b = (const float*)d_in[9];  u1_w = (const float*)d_in[10];
    u2_b = (const float*)d_in[11]; u2_w = (const float*)d_in[12];
    // sorted stage order: vim256, vim64d, vim64u ; leaves already canonical
    for (int blk = 0; blk < 2; ++blk)
      for (int j = 0; j < 13; ++j) {
        v256[blk].p[j] = (const float*)d_in[13 + blk * 13 + j];
        v64d[blk].p[j] = (const float*)d_in[39 + blk * 13 + j];
        v64u[blk].p[j] = (const float*)d_in[65 + blk * 13 + j];
      }
  } else {
    ci_w = (const float*)d_in[1];  ci_b = (const float*)d_in[2];
    d1_w = (const float*)d_in[3];  d1_b = (const float*)d_in[4];
    d2_w = (const float*)d_in[5];  d2_b = (const float*)d_in[6];
    u1_w = (const float*)d_in[7];  u1_b = (const float*)d_in[8];
    u2_w = (const float*)d_in[9];  u2_b = (const float*)d_in[10];
    co_w = (const float*)d_in[11]; co_b = (const float*)d_in[12];
    // insertion stage order: vim64d, vim256, vim64u
    // insertion leaf j -> canonical slot:
    static const int m[13] = {7, 6, 10, 4, 8, 2, 12, 9, 3, 0, 1, 11, 5};
    for (int blk = 0; blk < 2; ++blk)
      for (int j = 0; j < 13; ++j) {
        v64d[blk].p[m[j]] = (const float*)d_in[13 + blk * 13 + j];
        v256[blk].p[m[j]] = (const float*)d_in[39 + blk * 13 + j];
        v64u[blk].p[m[j]] = (const float*)d_in[65 + blk * 13 + j];
      }
  }

  float* out_prompt = (float*)d_out;              // [8,256,32,32] = 2,097,152
  float* out_img = out_prompt + 2097152;          // [8,3,256,256] = 1,572,864

  // --- workspace arena (fixed offsets) ---
  char* wsp = (char*)d_ws;
  size_t off = 0;
  auto alloc = [&](size_t bytes) {
    void* p = wsp + off;
    off += (bytes + 255) & ~(size_t)255;
    return p;
  };
  float* img0 = (float*)alloc(8388608ull * 4);   // [8,16,256,256]
  float* seqA = (float*)alloc(2097152ull * 4);   // token-major sequence
  float* seqB = (float*)alloc(2097152ull * 4);
  float* lnb  = (float*)alloc(2097152ull * 4);
  float* xz   = (float*)alloc(8388608ull * 4);   // [tokens, 4*dim]
  float* xcv  = (float*)alloc(4194304ull * 4);
  float* dbc  = (float*)alloc(2097152ull * 4);
  float* dtb  = (float*)alloc(4194304ull * 4);
  float* yb   = (float*)alloc(4194304ull * 4);
  bf16_t* A16 = (bf16_t*)alloc(4194304ull * 2);
  bf16_t* W16 = (bf16_t*)alloc(524288ull * 2);

  // 1) conv_in 3x3 reflect: [8,3,256,256] -> img0 [8,16,256,256]
  long long t = 8LL * 16 * 256 * 256;
  k_conv3x3_reflect<<<g1(t), 256, 0, stream>>>(c_in, ci_w, ci_b, img0, 8, 3, 16, 256, 256, t);

  // 2) down1 k4s4 -> seqA [8,4096,64]
  t = 8LL * 4096 * 64;
  k_down1<<<g1(t), 256, 0, stream>>>(img0, d1_w, d1_b, seqA, t);

  // 3) vim64d (2 blocks)
  run_mamba(stream, v64d[0], seqA, 8, 4096, 64, 16, lnb, xz, xcv, dbc, dtb, yb, A16, W16);
  run_mamba(stream, v64d[1], seqA, 8, 4096, 64, 16, lnb, xz, xcv, dbc, dtb, yb, A16, W16);

  // 4) down2 k2s2 via im2col + WMMA GEMM -> seqB [8,1024,256]
  t = 8192LL * 256;
  k_im2col_down2<<<g1(t), 256, 0, stream>>>(seqA, A16, t);
  k_pack_down2_w<<<g1(t / 32), 256, 0, stream>>>(d2_w, W16, 256LL * 256);
  launch_gemm(stream, A16, W16, d2_b, seqB, 8192, 256, 256, 0, 0);

  // 5) vim256 (2 blocks) -> prompt
  run_mamba(stream, v256[0], seqB, 8, 1024, 256, 64, lnb, xz, xcv, dbc, dtb, yb, A16, W16);
  run_mamba(stream, v256[1], seqB, 8, 1024, 256, 64, lnb, xz, xcv, dbc, dtb, yb, A16, W16);

  // 6) prompt -> NCHW in d_out
  t = 8LL * 256 * 32 * 32;
  k_seq_to_nchw<<<g1(t), 256, 0, stream>>>(seqB, out_prompt, 256, 32, 32, t);

  // 7) up1 1x1 conv (WMMA GEMM, weight [co,ci] is already Bt layout) + shuffle r=2
  t = 8192LL * 256;
  k_f32_to_bf16_rows<<<g1(t), 256, 0, stream>>>(seqB, 256, A16, 256, t);
  k_f32_to_bf16_rows<<<g1(65536LL), 256, 0, stream>>>(u1_w, 256, W16, 256, 65536LL);
  launch_gemm(stream, A16, W16, u1_b, xz, 8192, 256, 256, 0, 0);
  t = 8LL * 4096 * 64;
  k_ps2_seq<<<g1(t), 256, 0, stream>>>(xz, seqA, t);

  // 8) vim64u (2 blocks)
  run_mamba(stream, v64u[0], seqA, 8, 4096, 64, 16, lnb, xz, xcv, dbc, dtb, yb, A16, W16);
  run_mamba(stream, v64u[1], seqA, 8, 4096, 64, 16, lnb, xz, xcv, dbc, dtb, yb, A16, W16);

  // 9) up2 1x1 conv (WMMA GEMM) + shuffle r=4 -> img0 [8,16,256,256]
  t = 32768LL * 64;
  k_f32_to_bf16_rows<<<g1(t), 256, 0, stream>>>(seqA, 64, A16, 64, t);
  k_f32_to_bf16_rows<<<g1(16384LL), 256, 0, stream>>>(u2_w, 64, W16, 64, 16384LL);
  launch_gemm(stream, A16, W16, u2_b, xz, 32768, 256, 64, 0, 0);
  t = 8LL * 16 * 256 * 256;
  k_ps4_nchw<<<g1(t), 256, 0, stream>>>(xz, img0, t);

  // 10) conv_out 3x3 reflect -> out_img [8,3,256,256]
  t = 8LL * 3 * 256 * 256;
  k_conv3x3_reflect<<<g1(t), 256, 0, stream>>>(img0, co_w, co_b, out_img, 8, 16, 3, 256, 256, t);
}
